// Capsule_14456859918499
// MI455X (gfx1250) — matrix operand order repaired and checked by reading, not compiled, not run on hardware
//
#include <hip/hip_runtime.h>
#include <hip/hip_bf16.h>
#include <math.h>

typedef __attribute__((ext_vector_type(16))) _Float16 v16h;
typedef __attribute__((ext_vector_type(8)))  _Float16 v8h;
typedef __attribute__((ext_vector_type(8)))  float    v8f;
typedef __attribute__((ext_vector_type(4)))  unsigned int u32x4;
typedef __attribute__((ext_vector_type(8)))  int      i32x8;
typedef __attribute__((ext_vector_type(4)))  int      i32x4;

#define BATCH 32
#define SEQ   128
#define OUT_N 16
#define OUT_D 512
#define KDIM  1024
#define NDIM  8192   // OUT_N * OUT_D
#define MDIM  4096   // BATCH * SEQ
#define KCHUNK 128   // K elements staged per TDM transfer
#define BROW  136    // 128 data halfs + 8 pad halfs (TDM pad: 4 DWORDs / 64 DWORDs)

#if __has_builtin(__builtin_amdgcn_tanhf)
#define FAST_TANH(x) __builtin_amdgcn_tanhf(x)
#elif __has_builtin(__builtin_amdgcn_tanh_f32)
#define FAST_TANH(x) __builtin_amdgcn_tanh_f32(x)
#else
#define FAST_TANH(x) tanhf(x)
#endif

// ---------------------------------------------------------------- converts --
__global__ void cvt_x_kernel(const float* __restrict__ x,
                             _Float16* __restrict__ xh, int n) {
    int i = blockIdx.x * blockDim.x + threadIdx.x;
    if (i < n) xh[i] = (_Float16)x[i];
}

// W (K=1024, N=8192) f32 row-major  ->  wt (N, K) f16 row-major
__global__ void transW_kernel(const float* __restrict__ W,
                              _Float16* __restrict__ wt) {
    __shared__ _Float16 tile[32][33];
    const int n0 = blockIdx.x * 32;
    const int k0 = blockIdx.y * 32;
    const int tx = threadIdx.x, ty = threadIdx.y;
    tile[ty][tx] = (_Float16)W[(size_t)(k0 + ty) * NDIM + n0 + tx];
    __syncthreads();
    wt[(size_t)(n0 + ty) * KDIM + k0 + tx] = tile[tx][ty];
}

// --------------------------------------------------------- TDM tile stage ---
// Stage a 128(row) x 128(K) f16 tile of wt (row stride KDIM) into LDS at
// ldsOff, with LDS padding of 4 DWORDs every 64 DWORDs (row stride 272 B).
__device__ inline void tdm_stage_b(const _Float16* gptr, unsigned ldsOff) {
    unsigned long long ga = (unsigned long long)(uintptr_t)gptr;
    u32x4 g0;
    g0[0] = 1u;                                    // count=1, user descriptor
    g0[1] = ldsOff;                                // lds_addr (bytes)
    g0[2] = (unsigned)(ga & 0xffffffffu);          // global_addr[31:0]
    g0[3] = (unsigned)((ga >> 32) & 0x01ffffffu)   // global_addr[56:32]
            | 0x80000000u;                         // type = 2 ("image")
    i32x8 g1;
    g1[0] = (int)((1u << 16)                       // data_size = 2 bytes
                  | (1u << 20)                     // pad_enable
                  | (5u << 22)                     // pad_interval: 64 DWORDs
                  | (3u << 25));                   // pad_amount:   4 DWORDs
    g1[1] = (int)(((unsigned)KDIM & 0xffffu) << 16);   // tensor_dim0[15:0]
    g1[2] = (int)(((unsigned)NDIM & 0xffffu) << 16);   // dim0[31:16]=0 | tensor_dim1[15:0]
    g1[3] = (int)((unsigned)KCHUNK << 16);             // dim1[31:16]=0 | tile_dim0=128
    g1[4] = (int)(128u);                               // tile_dim1=128, tile_dim2=0
    g1[5] = (int)KDIM;                                 // tensor_dim0_stride[31:0]
    g1[6] = 0;                                         // stride[47:32]=0, dim1_stride lo=0
    g1[7] = 0;
    i32x4 z = {0, 0, 0, 0};
#if defined(__clang_major__) && (__clang_major__ >= 23)
    i32x8 z8 = {0, 0, 0, 0, 0, 0, 0, 0};
    __builtin_amdgcn_tensor_load_to_lds(g0, g1, z, z, z8, 0);
#else
    __builtin_amdgcn_tensor_load_to_lds(g0, g1, z, z, 0);
#endif
}

// ------------------------------------------------------------- WMMA GEMM ---
// U[m][n] = tanh( sum_k xh[m][k] * wt[n][k] ), stored f16.
// 4 waves / block; each wave computes a 64x64 tile as 4x4 WMMA accumulators.
// B operand (wt) is double-buffered through LDS via the Tensor Data Mover.
__global__ void __launch_bounds__(128)
gemm_tanh_kernel(const _Float16* __restrict__ xh,
                 const _Float16* __restrict__ wt,
                 _Float16* __restrict__ U) {
    __shared__ _Float16 bsm[2][128][BROW];        // 2 x 34 KB

    const int lane    = threadIdx.x & 31;
    const int wv      = threadIdx.x >> 5;          // 0..3
    const int M0      = blockIdx.x * 128 + (wv >> 1) * 64;
    const int NB0     = blockIdx.y * 128;          // block's N origin
    const int nLoc0   = (wv & 1) * 64;             // wave's N origin inside block
    const int halfSel = lane >> 4;                 // 0 | 1
    const int rowInT  = lane & 15;
    const int aBase   = halfSel * 8;   // A: lanes 0-15 -> K {0..7,16..23}; 16-31 -> {8..15,24..31}
    const int bBase   = halfSel * 8;   // B: lanes 0-15 -> K {0..7,16..23}; 16-31 -> {8..15,24..31}

    v8f acc[4][4] = {};

    // prime buffer 0 with the first K-chunk of this block's 128 wt rows
    if (wv == 0)
        tdm_stage_b(wt + (size_t)NB0 * KDIM,
                    (unsigned)(uintptr_t)&bsm[0][0][0]);

#pragma unroll 1
    for (int kc = 0; kc < KDIM; kc += KCHUNK) {
        const int buf = (kc >> 7) & 1;
        if (wv == 0) __builtin_amdgcn_s_wait_tensorcnt(0);
        __syncthreads();                           // chunk `buf` ready; prev reads done
        if (wv == 0 && (kc + KCHUNK) < KDIM)
            tdm_stage_b(wt + (size_t)NB0 * KDIM + kc + KCHUNK,
                        (unsigned)(uintptr_t)&bsm[buf ^ 1][0][0]);

#pragma unroll
        for (int kk = 0; kk < KCHUNK; kk += 32) {
            const int ks = kc + kk;
            v16h a[4], b[4];
#pragma unroll
            for (int mi = 0; mi < 4; ++mi) {
                const _Float16* p = xh + (size_t)(M0 + mi * 16 + rowInT) * KDIM + ks + aBase;
                union { v16h v; v8h h[2]; } u;
                u.h[0] = *(const v8h*)p;           // K offsets base+0..7
                u.h[1] = *(const v8h*)(p + 16);    // K offsets base+16..23
                a[mi] = u.v;
            }
#pragma unroll
            for (int ni = 0; ni < 4; ++ni) {
                const _Float16* p = &bsm[buf][nLoc0 + ni * 16 + rowInT][kk + bBase];
                union { v16h v; v8h h[2]; } u;
                u.h[0] = *(const v8h*)p;           // K offsets base+0..7
                u.h[1] = *(const v8h*)(p + 16);    // K offsets base+16..23
                b[ni] = u.v;
            }
#pragma unroll
            for (int mi = 0; mi < 4; ++mi)
#pragma unroll
                for (int ni = 0; ni < 4; ++ni)
                    acc[mi][ni] = __builtin_amdgcn_wmma_f32_16x16x32_f16(
                        false, a[mi], false, b[ni], (short)0, acc[mi][ni], false, false);
        }
        __syncthreads();                           // done reading `buf` this round
    }

    // C/D layout: lane 0-15: VGPR v -> M=v ; lane 16-31: M=v+8 ; N = lane&15
#pragma unroll
    for (int mi = 0; mi < 4; ++mi)
#pragma unroll
        for (int ni = 0; ni < 4; ++ni) {
#pragma unroll
            for (int v = 0; v < 8; ++v) {
                const int row = M0 + mi * 16 + v + halfSel * 8;
                const int col = NB0 + nLoc0 + ni * 16 + rowInT;
                U[(size_t)row * NDIM + col] = (_Float16)FAST_TANH(acc[mi][ni][v]);
            }
        }
}

// --------------------------------------------------------------- routing ---
__global__ void zeroB_kernel(float* __restrict__ B, int n) {
    int i = blockIdx.x * blockDim.x + threadIdx.x;
    if (i < n) B[i] = 0.f;
}

// C = softmax(B, axis=n); also snapshot C_ret (every iter) and B_logits (last iter)
__global__ void route_softmax_kernel(const float* __restrict__ B,
                                     float* __restrict__ C,
                                     float* __restrict__ Cret,
                                     float* __restrict__ Blog) {
    int idx = blockIdx.x * blockDim.x + threadIdx.x;   // (b*s) in [0,4096)
    if (idx >= MDIM) return;
    const float* br = B + (size_t)idx * OUT_N;
    float m = br[0];
#pragma unroll
    for (int n = 1; n < OUT_N; ++n) m = fmaxf(m, br[n]);
    float e[OUT_N]; float sum = 0.f;
#pragma unroll
    for (int n = 0; n < OUT_N; ++n) { e[n] = expf(br[n] - m); sum += e[n]; }
    const float inv = 1.f / sum;
#pragma unroll
    for (int n = 0; n < OUT_N; ++n) {
        const float c = e[n] * inv;
        C[(size_t)idx * OUT_N + n]    = c;
        Cret[(size_t)idx * OUT_N + n] = c;
        if (Blog) Blog[(size_t)idx * OUT_N + n] = br[n];
    }
}

// S[b,n,:] = sum_s C[b,s,n]*U[b,s,n,:]  ;  V = S * rsqrt(||S||^2 + eps)
// one block (256 thr) per (b,n); each thread owns d = t and d = t+256
__global__ void __launch_bounds__(256)
route_sv_kernel(const _Float16* __restrict__ U, const float* __restrict__ C,
                float* __restrict__ Vbuf, float* __restrict__ Vout) {
    const int b = blockIdx.x >> 4;
    const int n = blockIdx.x & 15;
    const int t = threadIdx.x;
    float s0 = 0.f, s1 = 0.f;
    const _Float16* ub = U + (size_t)b * SEQ * NDIM + (size_t)n * OUT_D;
    const float*    cb = C + (size_t)b * SEQ * OUT_N + n;
    for (int s = 0; s < SEQ; ++s) {
        const float c = cb[(size_t)s * OUT_N];
        const _Float16* up = ub + (size_t)s * NDIM;
        __builtin_prefetch(up + NDIM, 0, 3);       // global_prefetch: next seq row
        s0 += c * (float)up[t];
        s1 += c * (float)up[t + 256];
    }
    float nrm = s0 * s0 + s1 * s1;
#pragma unroll
    for (int off = 16; off > 0; off >>= 1) nrm += __shfl_xor(nrm, off, 32);
    __shared__ float wsum[8];
    if ((t & 31) == 0) wsum[t >> 5] = nrm;
    __syncthreads();
    float tot = 0.f;
#pragma unroll
    for (int w = 0; w < 8; ++w) tot += wsum[w];
    const float inv = rsqrtf(tot + 1e-7f);
    const float v0 = s0 * inv, v1 = s1 * inv;
    const size_t o = ((size_t)b * OUT_N + n) * OUT_D;
    Vbuf[o + t] = v0; Vbuf[o + t + 256] = v1;
    if (Vout) { Vout[o + t] = v0; Vout[o + t + 256] = v1; }
}

// B[b,s,n] += dot(U[b,s,n,:], V[b,n,:])  — one wave per (b,s,n)
__global__ void __launch_bounds__(256)
route_bupd_kernel(const _Float16* __restrict__ U, const float* __restrict__ Vbuf,
                  float* __restrict__ B) {
    const int gid  = blockIdx.x * blockDim.x + threadIdx.x;
    const int lane = gid & 31;
    const int gw   = gid >> 5;      // global wave id, [0, 65536)
    const int n    = gw & 15;
    const int bs   = gw >> 4;       // [0, 4096)
    const int b    = bs >> 7;
    const _Float16* up = U + (size_t)bs * NDIM + (size_t)n * OUT_D;
    const float*    vp = Vbuf + ((size_t)b * OUT_N + n) * OUT_D;
    float acc = 0.f;
    const int base = lane * 16;     // 16 contiguous elems per lane (512/32)
#pragma unroll
    for (int i = 0; i < 16; ++i) acc += (float)up[base + i] * vp[base + i];
#pragma unroll
    for (int off = 16; off > 0; off >>= 1) acc += __shfl_xor(acc, off, 32);
    if (lane == 0) B[(size_t)bs * OUT_N + n] += acc;
}

// ------------------------------------------------------------------ host ---
extern "C" void kernel_launch(void* const* d_in, const int* in_sizes, int n_in,
                              void* d_out, int out_size, void* d_ws, size_t ws_size,
                              hipStream_t stream) {
    const float* x = (const float*)d_in[0];   // (32,128,1024) f32
    const float* W = (const float*)d_in[1];   // (1024,8192)   f32
    float* out = (float*)d_out;
    char*  ws  = (char*)d_ws;

    // workspace layout (bytes)
    _Float16* xh = (_Float16*)(ws);                              //  8 MB
    _Float16* wt = (_Float16*)(ws + (size_t)8  * 1024 * 1024);   // 16 MB
    _Float16* U  = (_Float16*)(ws + (size_t)24 * 1024 * 1024);   // 64 MB (caps_uhat f16)
    float* Bbuf  = (float*)   (ws + (size_t)88 * 1024 * 1024);   // 256 KB
    float* Cbuf  = Bbuf + MDIM * OUT_N;                          // 256 KB
    float* Vbuf  = Cbuf + MDIM * OUT_N;                          //   1 MB

    // outputs flat in return order: V (262144), C_ret (3*65536), B_logits (65536)
    float* Vout = out;
    float* Cret = out + BATCH * OUT_N * OUT_D;
    float* Blog = Cret + 3 * MDIM * OUT_N;

    cvt_x_kernel<<<(MDIM * KDIM) / 256, 256, 0, stream>>>(x, xh, MDIM * KDIM);
    transW_kernel<<<dim3(NDIM / 32, KDIM / 32), dim3(32, 32), 0, stream>>>(W, wt);
    gemm_tanh_kernel<<<dim3(MDIM / 128, NDIM / 128), 128, 0, stream>>>(xh, wt, U);
    zeroB_kernel<<<(MDIM * OUT_N) / 256, 256, 0, stream>>>(Bbuf, MDIM * OUT_N);

    for (int it = 0; it < 3; ++it) {
        route_softmax_kernel<<<MDIM / 256, 256, 0, stream>>>(
            Bbuf, Cbuf, Cret + it * MDIM * OUT_N, (it == 2) ? Blog : nullptr);
        route_sv_kernel<<<BATCH * OUT_N, 256, 0, stream>>>(
            U, Cbuf, Vbuf, (it == 2) ? Vout : nullptr);
        route_bupd_kernel<<<(MDIM * OUT_N * 32) / 256, 256, 0, stream>>>(U, Vbuf, Bbuf);
    }
}
